// GroupedQueryAttention_85306640433858
// MI455X (gfx1250) — matrix operand order repaired and checked by reading, not compiled, not run on hardware
//
#include <hip/hip_runtime.h>

#define HEADS 32
#define KVH   8
#define DHEAD 128
#define HID   4096

typedef __attribute__((ext_vector_type(16))) __bf16 v16bf;
typedef __attribute__((ext_vector_type(8)))  __bf16 v8bf;
typedef __attribute__((ext_vector_type(8)))  float  v8f;
typedef __attribute__((ext_vector_type(4)))  unsigned int v4u;
typedef __attribute__((ext_vector_type(8)))  int v8i;
typedef __attribute__((ext_vector_type(4)))  int v4i;

static __device__ __forceinline__ v16bf cat16(v8bf lo, v8bf hi) {
  return __builtin_shufflevector(lo, hi, 0,1,2,3,4,5,6,7,8,9,10,11,12,13,14,15);
}
static __device__ __forceinline__ v8f wmma_bf16(v16bf a, v16bf b, v8f c) {
  // 8 args: (neg_a, A, neg_b, B, c_mod, C, reuse_a, reuse_b)
  return __builtin_amdgcn_wmma_f32_16x16x32_bf16(false, a, false, b, (short)0, c, false, false);
}

// Low 32 bits of a generic pointer to __shared__ data == LDS byte offset
// (ISA 10.2: the LDS aperture uses the low 32 address bits).
static __device__ __forceinline__ unsigned lds_byte_off(const void* p) {
  return (unsigned)(unsigned long long)(size_t)p;
}

// --------------------------------------------------------------------------
// TDM: issue an async 2D bf16 tile load Global->LDS.
//   tile rows = tile_d1, row length = tile_d0 elements (256B), row stride in
//   memory = stride0 elements. LDS pad-on-load: +4 dwords every 64 dwords
//   (one row) -> 272B LDS row stride, bank-conflict-free ds_load_b128 reads.
// --------------------------------------------------------------------------
static __device__ __forceinline__ void tdm_load_tile_2d(
    unsigned lds_off, const void* gptr,
    unsigned tensor_d0, unsigned tensor_d1,
    unsigned long long stride0_elems,
    unsigned tile_d0, unsigned tile_d1)
{
  unsigned long long ga = (unsigned long long)(size_t)gptr;
  v4u g0;
  g0[0] = 1u;                                   // count=1, user descriptor
  g0[1] = lds_off;                              // LDS byte address
  g0[2] = (unsigned)ga;                         // global_addr[31:0]
  g0[3] = (unsigned)(ga >> 32) | (2u << 30);    // global_addr[56:32] | type=2
  v8i g1;
  unsigned c0 = (1u << 16)                      // data_size: 2 bytes
              | (1u << 20)                      // pad_enable
              | (5u << 22)                      // pad_interval: 64 dwords
              | (3u << 25);                     // pad_amount: 4 dwords (16B)
  g1[0] = (int)c0;
  g1[1] = (int)((tensor_d0 & 0xFFFFu) << 16);                         // dim0[15:0]
  g1[2] = (int)(((tensor_d0 >> 16) & 0xFFFFu) | ((tensor_d1 & 0xFFFFu) << 16));
  g1[3] = (int)(((tensor_d1 >> 16) & 0xFFFFu) | (tile_d0 << 16));
  g1[4] = (int)tile_d1;                                               // tile_dim1 | tile_dim2=0
  g1[5] = (int)(unsigned)(stride0_elems & 0xFFFFFFFFull);             // stride0[31:0]
  g1[6] = (int)(unsigned)((stride0_elems >> 32) & 0xFFFFull);         // stride0[47:32]
  g1[7] = 0;
  v4i z4 = {0, 0, 0, 0};
  v8i z8 = {0, 0, 0, 0, 0, 0, 0, 0};
  // amdgpu-toolchain (clang-23) 6-arg form: (g0, g1, g2, g3, g4, cpol)
  __builtin_amdgcn_tensor_load_to_lds(g0, g1, z4, z4, z8, 0);
}

// --------------------------------------------------------------------------
// Generic GEMM: C[M,N] = A[M,K](f32) * B[K,N](f32) + bias(f32), bf16 WMMA core
// Block: 256 thr (8 waves). Tile 128x128, K-step 32. Wave = 32x64 sub-tile.
// --------------------------------------------------------------------------
#define LDK 40  // padded LDS K-stride (elements) -> 80B rows, conflict-friendly

__global__ __launch_bounds__(256)
void gemm_bf16_wmma(const float* __restrict__ A, const float* __restrict__ B,
                    const float* __restrict__ bias, float* __restrict__ C,
                    int M, int N, int K)
{
  __shared__ __bf16 As[128 * LDK];  // [m][k]
  __shared__ __bf16 Bs[128 * LDK];  // [n][k]  (B transposed into LDS)

  const int tid  = threadIdx.x;
  const int lane = tid & 31;
  const int wid  = tid >> 5;
  const int wr   = wid & 3;     // 4 row-groups of 32
  const int wc   = wid >> 2;    // 2 col-groups of 64
  const int l16  = lane & 15;
  const int hv   = lane >> 4;   // half-wave select
  const int bm   = blockIdx.y * 128;
  const int bn   = blockIdx.x * 128;

  // per-thread staging coordinates (q = tid + j*256)
  const int ar = tid >> 3;             // A row 0..127   (j adds 32)
  const int ak = (tid & 7) << 2;       // A k col 0..28
  const int bk = tid >> 5;             // B k row 0..31  (j adds 8)
  const int bn4 = (tid & 31) << 2;     // B n col 0..124

  v8f acc[2][4];
#pragma unroll
  for (int i = 0; i < 2; ++i)
#pragma unroll
    for (int j = 0; j < 4; ++j)
#pragma unroll
      for (int r = 0; r < 8; ++r) acc[i][j][r] = 0.f;

  for (int k0 = 0; k0 < K; k0 += 32) {
    // ---- issue ALL 8 tile-fill loads first (independent, clause-able) ----
    float4 fa[4], fb[4];
#pragma unroll
    for (int j = 0; j < 4; ++j) {
      const int r = ar + j * 32;                     // row 0..127
      fa[j] = *(const float4*)(A + (size_t)(bm + r) * K + (k0 + ak));
    }
#pragma unroll
    for (int j = 0; j < 4; ++j) {
      const int kr = bk + j * 8;                     // k row 0..31
      fb[j] = *(const float4*)(B + (size_t)(k0 + kr) * N + (bn + bn4));
    }
    // ---- convert f32 -> bf16 and store to LDS ----
#pragma unroll
    for (int j = 0; j < 4; ++j) {
      const int r = ar + j * 32;
      __bf16* d = &As[r * LDK + ak];
      d[0] = (__bf16)fa[j].x; d[1] = (__bf16)fa[j].y;
      d[2] = (__bf16)fa[j].z; d[3] = (__bf16)fa[j].w;
    }
#pragma unroll
    for (int j = 0; j < 4; ++j) {
      const int kr = bk + j * 8;
      Bs[(bn4 + 0) * LDK + kr] = (__bf16)fb[j].x;
      Bs[(bn4 + 1) * LDK + kr] = (__bf16)fb[j].y;
      Bs[(bn4 + 2) * LDK + kr] = (__bf16)fb[j].z;
      Bs[(bn4 + 3) * LDK + kr] = (__bf16)fb[j].w;
    }
    // prefetch next K-step while this tile is consumed
    if (k0 + 32 < K) {
      __builtin_prefetch(A + (size_t)(bm + ar) * K + (k0 + 32 + ak), 0, 3);
      __builtin_prefetch(B + (size_t)(k0 + 32 + bk) * N + (bn + bn4), 0, 3);
    }
    __syncthreads();

    // A frag: lane<16 -> row=l16, K0..15; lane>=16 -> K16..31
    v16bf afr[2];
#pragma unroll
    for (int mi = 0; mi < 2; ++mi) {
      const __bf16* p = &As[(wr * 32 + mi * 16 + l16) * LDK + hv * 16];
      afr[mi] = cat16(*(const v8bf*)p, *(const v8bf*)(p + 8));
    }
    // B frag: lane<16 -> col=l16, K0..15; lane>=16 -> K16..31
    v16bf bfr[4];
#pragma unroll
    for (int ni = 0; ni < 4; ++ni) {
      const __bf16* p = &Bs[(wc * 64 + ni * 16 + l16) * LDK + hv * 16];
      bfr[ni] = cat16(*(const v8bf*)p, *(const v8bf*)(p + 8));
    }
#pragma unroll
    for (int mi = 0; mi < 2; ++mi)
#pragma unroll
      for (int ni = 0; ni < 4; ++ni)
        acc[mi][ni] = wmma_bf16(afr[mi], bfr[ni], acc[mi][ni]);
    __syncthreads();
  }

  // epilogue: C-layout elem r -> row = r + 8*hv, col = l16
#pragma unroll
  for (int mi = 0; mi < 2; ++mi)
#pragma unroll
    for (int ni = 0; ni < 4; ++ni) {
      const int col = bn + wc * 64 + ni * 16 + l16;
      const float bv = bias[col];
#pragma unroll
      for (int r = 0; r < 8; ++r) {
        const int row = bm + wr * 32 + mi * 16 + hv * 8 + r;
        C[(size_t)row * N + col] = acc[mi][ni][r] + bv;
      }
    }
}

// --------------------------------------------------------------------------
// RoPE + pack kernels (fp32 -> bf16, layouts for the attention WMMAs)
// --------------------------------------------------------------------------
__global__ void rope_pack_q(const float* __restrict__ qf, __bf16* __restrict__ qb, int S)
{
  int i = blockIdx.x * 256 + threadIdx.x;
  if (i >= S * HEADS * (DHEAD / 2)) return;
  int d = i & 63;
  int h = (i >> 6) & (HEADS - 1);
  int s = i >> 11;
  const float* src = qf + (size_t)s * (HEADS * DHEAD) + h * DHEAD;
  float x1 = src[d], x2 = src[d + 64];
  float ang = (float)s * __expf(-(float)d * 0.14391157f);  // 1/theta^(d/64)
  float sn, cs; __sincosf(ang, &sn, &cs);
  const float sc = 0.08838834764831845f;                   // 1/sqrt(128)
  __bf16* dst = qb + ((size_t)h * S + s) * DHEAD;
  dst[d]      = (__bf16)((x1 * cs - x2 * sn) * sc);
  dst[d + 64] = (__bf16)((x1 * sn + x2 * cs) * sc);
}

__global__ void rope_pack_k(const float* __restrict__ kf, __bf16* __restrict__ kb, int S)
{
  int i = blockIdx.x * 256 + threadIdx.x;
  if (i >= S * KVH * (DHEAD / 2)) return;
  int d = i & 63;
  int g = (i >> 6) & (KVH - 1);
  int s = i >> 9;
  const float* src = kf + (size_t)s * (KVH * DHEAD) + g * DHEAD;
  float x1 = src[d], x2 = src[d + 64];
  float ang = (float)s * __expf(-(float)d * 0.14391157f);
  float sn, cs; __sincosf(ang, &sn, &cs);
  __bf16* dst = kb + ((size_t)g * S + s) * DHEAD;
  dst[d]      = (__bf16)(x1 * cs - x2 * sn);
  dst[d + 64] = (__bf16)(x1 * sn + x2 * cs);
}

__global__ void pack_vT(const float* __restrict__ vf, __bf16* __restrict__ vT, int S)
{
  int i = blockIdx.x * 256 + threadIdx.x;
  if (i >= S * KVH * DHEAD) return;
  int d = i & (DHEAD - 1);
  int g = (i >> 7) & (KVH - 1);
  int s = i >> 10;
  vT[((size_t)g * DHEAD + d) * S + s] = (__bf16)vf[(size_t)s * (KVH * DHEAD) + g * DHEAD + d];
}

// --------------------------------------------------------------------------
// Flash-style causal GQA attention. Block = (128 q rows, 1 head), 8 waves,
// each wave owns 16 q rows. K/V tiles staged via TDM (double-buffered),
// online softmax with wave32 shfl_xor reductions.
// --------------------------------------------------------------------------
#define LDT 136   // padded LDS row stride for TDM tiles (elements) = 272B

__global__ __launch_bounds__(256)
void attn_fwd(const __bf16* __restrict__ qb, const __bf16* __restrict__ kb,
              const __bf16* __restrict__ vT, float* __restrict__ ctx, int S)
{
  __shared__ __bf16 Kt[2][128 * LDT];   // K tile: row t-local, 272B stride
  __shared__ __bf16 Vt[2][128 * LDT];   // V^T tile: row d, 272B stride
  __shared__ __bf16 Plds[8][16 * LDT];  // per-wave P slab

  const int lane  = threadIdx.x & 31;
  const int wid   = threadIdx.x >> 5;
  const int l16   = lane & 15;
  const int hv    = lane >> 4;
  const int h     = blockIdx.y;
  const int g     = h >> 2;              // R = H/G = 4
  const int qbase = blockIdx.x * 128;
  const int srow0 = qbase + wid * 16;

  const __bf16* qh = qb + (size_t)h * S * DHEAD;
  const __bf16* kh = kb + (size_t)g * S * DHEAD;
  const __bf16* vh = vT + (size_t)g * DHEAD * S;

  float mrow[8], lrow[8];
  v8f oacc[8];
#pragma unroll
  for (int r = 0; r < 8; ++r) { mrow[r] = -3.0e38f; lrow[r] = 0.f; }
#pragma unroll
  for (int j = 0; j < 8; ++j)
#pragma unroll
    for (int r = 0; r < 8; ++r) oacc[j][r] = 0.f;

  __bf16* pw = &Plds[wid][0];

  // prologue: DMA first K/V tiles into buffer 0
  if (wid == 0) {
    tdm_load_tile_2d(lds_byte_off(&Kt[0][0]), kh, 128, 128, 128, 128, 128);
    tdm_load_tile_2d(lds_byte_off(&Vt[0][0]), vh, 128, 128, (unsigned long long)S, 128, 128);
  }

  int buf = 0;
  for (int t0 = 0; t0 <= qbase + 127; t0 += 128) {   // causal: tiles up to diag
    const bool more = (t0 < qbase);
    if (wid == 0) {
      if (more) {
        const __bf16* knext = kh + (size_t)(t0 + 128) * DHEAD;
        const __bf16* vnext = vh + (t0 + 128);
        tdm_load_tile_2d(lds_byte_off(&Kt[buf ^ 1][0]), knext, 128, 128, 128, 128, 128);
        tdm_load_tile_2d(lds_byte_off(&Vt[buf ^ 1][0]), vnext, 128, 128, (unsigned long long)S, 128, 128);
        __builtin_amdgcn_s_wait_tensorcnt(2);   // current buffer's 2 DMAs done
      } else {
        __builtin_amdgcn_s_wait_tensorcnt(0);
      }
    }
    __syncthreads();   // publish TDM-written LDS to all waves

    const __bf16* ktile = &Kt[buf][0];
    const __bf16* vtile = &Vt[buf][0];

    v8f sacc[8];
#pragma unroll
    for (int j = 0; j < 8; ++j)
#pragma unroll
      for (int r = 0; r < 8; ++r) sacc[j][r] = 0.f;

    // S-tile = Q(16xD) * K^T(Dx128): B frags from the LDS K tile
#pragma unroll
    for (int d0 = 0; d0 < DHEAD; d0 += 32) {
      const __bf16* qrow = qh + (size_t)(srow0 + l16) * DHEAD + d0 + hv * 16;
      v16bf a = cat16(*(const v8bf*)qrow, *(const v8bf*)(qrow + 8));
#pragma unroll
      for (int j = 0; j < 8; ++j) {
        const __bf16* krow = ktile + (j * 16 + l16) * LDT + d0 + hv * 16;
        v16bf b = cat16(*(const v8bf*)krow, *(const v8bf*)(krow + 8));
        sacc[j] = wmma_bf16(a, b, sacc[j]);
      }
    }

    // causal mask + online softmax update (rows live in 16-lane halves)
#pragma unroll
    for (int r = 0; r < 8; ++r) {
      const int row = srow0 + hv * 8 + r;
      float mx = -3.0e38f;
#pragma unroll
      for (int j = 0; j < 8; ++j) {
        const int t = t0 + j * 16 + l16;
        float v = sacc[j][r];
        if (t > row) v = -3.0e38f;
        sacc[j][r] = v;
        mx = fmaxf(mx, v);
      }
#pragma unroll
      for (int off = 1; off < 16; off <<= 1) mx = fmaxf(mx, __shfl_xor(mx, off, 32));
      const float mn    = fmaxf(mrow[r], mx);
      const float alpha = __expf(mrow[r] - mn);
      mrow[r] = mn;
      float rs = 0.f;
#pragma unroll
      for (int j = 0; j < 8; ++j) {
        float p = __expf(sacc[j][r] - mn);
        sacc[j][r] = p;
        rs += p;
      }
#pragma unroll
      for (int off = 1; off < 16; off <<= 1) rs += __shfl_xor(rs, off, 32);
      lrow[r] = lrow[r] * alpha + rs;
#pragma unroll
      for (int j = 0; j < 8; ++j) oacc[j][r] *= alpha;
    }

    // stage P (16x128) through LDS to reshape C-layout -> A-fragment layout
#pragma unroll
    for (int j = 0; j < 8; ++j)
#pragma unroll
      for (int r = 0; r < 8; ++r)
        pw[(hv * 8 + r) * LDT + j * 16 + l16] = (__bf16)sacc[j][r];
    __builtin_amdgcn_wave_barrier();   // per-wave LDS ops complete in order

    // O += P(16x128) * V(128xD): B frags from the LDS V^T tile
#pragma unroll
    for (int kk = 0; kk < 128; kk += 32) {
      const __bf16* prow = pw + l16 * LDT + kk + hv * 16;
      v16bf a = cat16(*(const v8bf*)prow, *(const v8bf*)(prow + 8));
#pragma unroll
      for (int j = 0; j < 8; ++j) {
        const __bf16* vrow = vtile + (j * 16 + l16) * LDT + kk + hv * 16;
        v16bf b = cat16(*(const v8bf*)vrow, *(const v8bf*)(vrow + 8));
        oacc[j] = wmma_bf16(a, b, oacc[j]);
      }
    }
    __syncthreads();   // all waves done reading buf before it is re-filled
    buf ^= 1;
  }

  // normalize and scatter ctx (s, h*D + d) fp32 for the output projection
#pragma unroll
  for (int j = 0; j < 8; ++j)
#pragma unroll
    for (int r = 0; r < 8; ++r) {
      const int row = srow0 + hv * 8 + r;
      ctx[(size_t)row * (HEADS * DHEAD) + h * DHEAD + j * 16 + l16] =
          oacc[j][r] / lrow[r];
    }
}

// --------------------------------------------------------------------------
extern "C" void kernel_launch(void* const* d_in, const int* in_sizes, int n_in,
                              void* d_out, int out_size, void* d_ws, size_t ws_size,
                              hipStream_t stream)
{
  const float* hs = (const float*)d_in[0];
  const float* Wq = (const float*)d_in[1];
  const float* bq = (const float*)d_in[2];
  const float* Wk = (const float*)d_in[3];
  const float* bk = (const float*)d_in[4];
  const float* Wv = (const float*)d_in[5];
  const float* bv = (const float*)d_in[6];
  const float* Wo = (const float*)d_in[7];
  const float* bo = (const float*)d_in[8];
  float* out = (float*)d_out;

  const int S = in_sizes[0] / HID;                 // 2048
  const size_t SH = (size_t)S * HEADS * DHEAD;     // q/ctx elems
  const size_t SG = (size_t)S * KVH * DHEAD;       // k/v elems

  char* p = (char*)d_ws;
  float*  qf   = (float*)p;  p += SH * sizeof(float);
  float*  kf   = (float*)p;  p += SG * sizeof(float);
  float*  vf   = (float*)p;  p += SG * sizeof(float);
  float*  ctx  = (float*)p;  p += SH * sizeof(float);
  __bf16* qbuf = (__bf16*)p; p += SH * sizeof(__bf16);
  __bf16* kbuf = (__bf16*)p; p += SG * sizeof(__bf16);
  __bf16* vbuf = (__bf16*)p; p += SG * sizeof(__bf16);

  dim3 blk(256);
  // QKV projections
  gemm_bf16_wmma<<<dim3((HEADS * DHEAD) / 128, S / 128), blk, 0, stream>>>(
      hs, Wq, bq, qf, S, HEADS * DHEAD, HID);
  gemm_bf16_wmma<<<dim3((KVH * DHEAD) / 128, S / 128), blk, 0, stream>>>(
      hs, Wk, bk, kf, S, KVH * DHEAD, HID);
  gemm_bf16_wmma<<<dim3((KVH * DHEAD) / 128, S / 128), blk, 0, stream>>>(
      hs, Wv, bv, vf, S, KVH * DHEAD, HID);

  // RoPE + bf16 packing
  int tq = S * HEADS * (DHEAD / 2);
  rope_pack_q<<<(tq + 255) / 256, blk, 0, stream>>>(qf, qbuf, S);
  int tk = S * KVH * (DHEAD / 2);
  rope_pack_k<<<(tk + 255) / 256, blk, 0, stream>>>(kf, kbuf, S);
  int tv = S * KVH * DHEAD;
  pack_vT<<<(tv + 255) / 256, blk, 0, stream>>>(vf, vbuf, S);

  // causal GQA attention (TDM-staged K/V tiles)
  attn_fwd<<<dim3(S / 128, HEADS), blk, 0, stream>>>(qbuf, kbuf, vbuf, ctx, S);

  // output projection
  gemm_bf16_wmma<<<dim3(HID / 128, S / 128), blk, 0, stream>>>(
      ctx, Wo, bo, out, S, HID, HEADS * DHEAD);
}
